// ReplayCritic_54013508714832
// MI455X (gfx1250) — matrix operand order repaired and checked by reading, not compiled, not run on hardware
//
#include <hip/hip_runtime.h>

// Problem constants (match reference)
#define TT   2048
#define BB   64
#define HH   128
#define G4H  512                 // 4*H
#define TB   (TT * BB)           // 131072 rows

typedef __attribute__((ext_vector_type(16))) __bf16 v16bf;
typedef __attribute__((ext_vector_type(8)))  float  v8f;

union FragB { v16bf v; unsigned int u[8]; };
union FragC { v8f   v; float        f[8]; };

__device__ __forceinline__ unsigned short f2bf(float x) {
    unsigned int u = __float_as_uint(x);
    u += 0x7FFFu + ((u >> 16) & 1u);          // round-to-nearest-even
    return (unsigned short)(u >> 16);
}
__device__ __forceinline__ float bf2f(unsigned short b) {
    return __uint_as_float(((unsigned int)b) << 16);
}
__device__ __forceinline__ float sigmoidf_(float x) {
    return 1.0f / (1.0f + __expf(-x));
}
// Low 32 bits of a generic pointer into LDS == wave-relative LDS byte address.
__device__ __forceinline__ unsigned lds_addr_of(const void* p) {
    return (unsigned)(unsigned long long)p;
}

// ---------------------------------------------------------------------------
// Pack x0 = concat(map_features, positions), time-major, bf16: [T][B][128]
// ---------------------------------------------------------------------------
__global__ void pack_x0_kernel(const float* __restrict__ map,
                               const float* __restrict__ pos,
                               unsigned short* __restrict__ x0) {
    int idx = blockIdx.x * 256 + threadIdx.x;        // over TB*128
    int k = idx & 127;
    int tb = idx >> 7;
    int b = tb & 63;
    int t = tb >> 6;
    float v;
    if (k < 126) v = map[((size_t)b * TT + t) * 126 + k];
    else         v = pos[((size_t)b * TT + t) * 2 + (k - 126)];
    x0[idx] = f2bf(v);
}

// fp32 -> bf16 bulk convert
__global__ void cvt_bf16_kernel(const float* __restrict__ src,
                                unsigned short* __restrict__ dst, int n) {
    int i = blockIdx.x * 256 + threadIdx.x;
    if (i < n) dst[i] = f2bf(src[i]);
}

// bias = bih + bhh
__global__ void bias_kernel(const float* __restrict__ a,
                            const float* __restrict__ b,
                            float* __restrict__ out) {
    int i = blockIdx.x * 256 + threadIdx.x;
    if (i < G4H) out[i] = a[i] + b[i];
}

// ---------------------------------------------------------------------------
// xproj GEMM: out_bf16[M,512] = A_bf16[M,128] @ W_bf16[512,128]^T + bias
// Each wave computes one 16-row M-tile x 64 columns (4 N-tiles), K=128.
// ---------------------------------------------------------------------------
__global__ __launch_bounds__(256) void xproj_gemm_kernel(
    const unsigned short* __restrict__ a_bf,   // [M][128] bf16
    const unsigned short* __restrict__ w_bf,   // [512][128] bf16
    const float* __restrict__ bias,            // [512]
    unsigned short* __restrict__ out)          // [M][512] bf16
{
    const int lane = threadIdx.x & 31;
    const int wave = threadIdx.x >> 5;
    const int wt   = blockIdx.x * 8 + wave;
    const int mtile = wt >> 3;
    const int ngrp  = wt & 7;
    const int mrow = lane & 15;
    const int kh   = lane >> 4;
    const int row  = mtile * 16 + mrow;

    // per-lane K-pair offsets inside one 32-wide K chunk (documented A layout)
    int koff[8];
#pragma unroll
    for (int d = 0; d < 8; ++d)
        koff[d] = (d < 4) ? (8 * kh + 2 * d) : (16 + 8 * kh + 2 * (d - 4));

    FragB afrag[4];
#pragma unroll
    for (int kc = 0; kc < 4; ++kc)
#pragma unroll
        for (int d = 0; d < 8; ++d) {
            int k = kc * 32 + koff[d];
            afrag[kc].u[d] = *(const unsigned int*)(a_bf + (size_t)row * 128 + k);
        }

#pragma unroll
    for (int nt = 0; nt < 4; ++nt) {
        const int n = ngrp * 64 + nt * 16 + mrow;
        FragC acc;
#pragma unroll
        for (int i = 0; i < 8; ++i) acc.f[i] = 0.0f;
#pragma unroll
        for (int kc = 0; kc < 4; ++kc) {
            FragB bf;
#pragma unroll
            for (int d = 0; d < 8; ++d) {
                int k = kc * 32 + koff[d];  // B layout: lane=N, K-pairs (same order as A)
                bf.u[d] = *(const unsigned int*)(w_bf + (size_t)n * 128 + k);
            }
            acc.v = __builtin_amdgcn_wmma_f32_16x16x32_bf16(
                false, afrag[kc].v, false, bf.v, (short)0, acc.v, false, false);
        }
        const float bn = bias[n];
#pragma unroll
        for (int r = 0; r < 8; ++r) {
            int m = r + 8 * kh;                     // C layout: VGPR r -> row
            out[(size_t)(mtile * 16 + m) * G4H + n] = f2bf(acc.f[r] + bn);
        }
    }
}

// ---------------------------------------------------------------------------
// Persistent recurrent LSTM kernel: one workgroup owns 16 batch rows for all T.
//  * Whh held in registers as WMMA B-fragments (loaded once).
//  * Wave w computes gate columns {g*128 + w*16 .. +16} for g=i,f,g,o, so the
//    whole cell update stays in registers (c-state never leaves VGPRs).
//  * xproj tiles (16x512 bf16, contiguous 16KB) double-buffered in LDS via
//    async global->LDS loads, prefetched one timestep ahead (ASYNCcnt).
// ---------------------------------------------------------------------------
__global__ __launch_bounds__(256) void lstm_rec_kernel(
    const unsigned short* __restrict__ xproj,     // [T*64][512] bf16 (w/ bias)
    const unsigned short* __restrict__ whh_bf,    // [512][128] bf16
    unsigned short* __restrict__ hs_out,          // [T*64][128] bf16 or null
    float* __restrict__ hT_out)                   // [64][128] f32 or null
{
    __shared__ unsigned short hbuf[16][128];      // h_{t-1}, bf16 (4 KB)
    __shared__ unsigned short xpb[2][16][G4H];    // xproj tiles   (32 KB)

    const int tid  = threadIdx.x;
    const int lane = tid & 31;
    const int wave = tid >> 5;                    // 0..7
    const int batch0 = blockIdx.x * 16;
    const int mrow = lane & 15;
    const int kh   = lane >> 4;
    const int ucol = wave * 16 + mrow;            // this lane's hidden unit

    for (int i = tid; i < 16 * 128; i += 256)
        ((unsigned short*)hbuf)[i] = 0;

    int koff[8];
#pragma unroll
    for (int d = 0; d < 8; ++d)
        koff[d] = (d < 4) ? (8 * kh + 2 * d) : (16 + 8 * kh + 2 * (d - 4));

    // Load Whh B-fragments once; bfrag[g] covers gate-g columns of this wave.
    FragB bfrag[4][4];                            // [gate][kchunk]
#pragma unroll
    for (int g = 0; g < 4; ++g) {
        const int n = g * 128 + ucol;             // gate column (global)
#pragma unroll
        for (int kc = 0; kc < 4; ++kc)
#pragma unroll
            for (int d = 0; d < 8; ++d) {
                int k = kc * 32 + koff[d];
                bfrag[g][kc].u[d] =
                    *(const unsigned int*)(whh_bf + (size_t)n * 128 + k);
            }
    }

    float creg[8];                                // c-state: (row r+8*kh, ucol)
#pragma unroll
    for (int i = 0; i < 8; ++i) creg[i] = 0.0f;

    // ---- async prefetch of xproj tile for t=0 into xpb[0] ----
    const unsigned long long xbase =
        (unsigned long long)(xproj + (size_t)batch0 * G4H);   // t=0 tile start
    {
        unsigned long long ga = xbase + (unsigned long long)tid * 64ull;
        unsigned la = lds_addr_of(&xpb[0][0][0]) + (unsigned)tid * 64u;
        asm volatile("global_load_async_to_lds_b128 %0, %1, off"
                     :: "v"(la), "v"(ga) : "memory");
        asm volatile("global_load_async_to_lds_b128 %0, %1, off offset:16"
                     :: "v"(la), "v"(ga) : "memory");
        asm volatile("global_load_async_to_lds_b128 %0, %1, off offset:32"
                     :: "v"(la), "v"(ga) : "memory");
        asm volatile("global_load_async_to_lds_b128 %0, %1, off offset:48"
                     :: "v"(la), "v"(ga) : "memory");
    }
    asm volatile("s_wait_asynccnt 0x0" ::: "memory");
    __syncthreads();

    for (int t = 0; t < TT; ++t) {
        const int pb = t & 1;

        // prefetch tile for t+1 into the other buffer (overlaps this step)
        if (t + 1 < TT) {
            unsigned long long ga = xbase +
                ((unsigned long long)(t + 1) * BB * G4H + tid * 32ull) * 2ull;
            unsigned la = lds_addr_of(&xpb[pb ^ 1][0][0]) + (unsigned)tid * 64u;
            asm volatile("global_load_async_to_lds_b128 %0, %1, off"
                         :: "v"(la), "v"(ga) : "memory");
            asm volatile("global_load_async_to_lds_b128 %0, %1, off offset:16"
                         :: "v"(la), "v"(ga) : "memory");
            asm volatile("global_load_async_to_lds_b128 %0, %1, off offset:32"
                         :: "v"(la), "v"(ga) : "memory");
            asm volatile("global_load_async_to_lds_b128 %0, %1, off offset:48"
                         :: "v"(la), "v"(ga) : "memory");
        }

        // A fragments from h_{t-1} in LDS
        FragB afrag[4];
#pragma unroll
        for (int kc = 0; kc < 4; ++kc)
#pragma unroll
            for (int d = 0; d < 8; ++d)
                afrag[kc].u[d] =
                    *(const unsigned int*)&hbuf[mrow][kc * 32 + koff[d]];

        // 4 gate tiles: gv[g][r] = (h @ Whh^T)[m][g*128+ucol] + xproj
        float gv[4][8];
#pragma unroll
        for (int g = 0; g < 4; ++g) {
            FragC acc;
#pragma unroll
            for (int i = 0; i < 8; ++i) acc.f[i] = 0.0f;
#pragma unroll
            for (int kc = 0; kc < 4; ++kc)
                acc.v = __builtin_amdgcn_wmma_f32_16x16x32_bf16(
                    false, afrag[kc].v, false, bfrag[g][kc].v,
                    (short)0, acc.v, false, false);
#pragma unroll
            for (int r = 0; r < 8; ++r) {
                int m = r + 8 * kh;
                gv[g][r] = acc.f[r] + bf2f(xpb[pb][m][g * 128 + ucol]);
            }
        }

        // in-register cell update for (rows r+8*kh, unit ucol)
        float hv[8];
#pragma unroll
        for (int r = 0; r < 8; ++r) {
            float ig = sigmoidf_(gv[0][r]);
            float fg = sigmoidf_(gv[1][r]);
            float gg = tanhf(gv[2][r]);
            float og = sigmoidf_(gv[3][r]);
            float c  = fg * creg[r] + ig * gg;
            creg[r]  = c;
            hv[r]    = og * tanhf(c);
        }
#pragma unroll
        for (int r = 0; r < 8; ++r)
            hbuf[r + 8 * kh][ucol] = f2bf(hv[r]);
        if (hs_out) {
#pragma unroll
            for (int r = 0; r < 8; ++r) {
                int m = r + 8 * kh;
                hs_out[((size_t)t * BB + batch0 + m) * HH + ucol] = f2bf(hv[r]);
            }
        }

        // drain prefetch, then make hbuf + xpb[t+1] visible to all waves
        asm volatile("s_wait_asynccnt 0x0" ::: "memory");
        __syncthreads();
    }

    if (hT_out) {
        for (int i = tid; i < 16 * HH; i += 256) {
            int m = i >> 7, u = i & 127;
            hT_out[(size_t)(batch0 + m) * HH + u] = bf2f(hbuf[m][u]);
        }
    }
}

// ---------------------------------------------------------------------------
// MLP head: h = hT@W1^T+b1 -> LayerNorm -> LeakyReLU(0.2) -> @W2^T+b2
// ---------------------------------------------------------------------------
__global__ void head_kernel(const float* __restrict__ hT,
                            const float* __restrict__ W1,
                            const float* __restrict__ b1,
                            const float* __restrict__ ln_g,
                            const float* __restrict__ ln_b,
                            const float* __restrict__ W2,
                            const float* __restrict__ b2,
                            float* __restrict__ out) {
    int b = threadIdx.x;
    if (b >= BB) return;
    float h[64];
    for (int j = 0; j < 64; ++j) {
        float s = b1[j];
        for (int k = 0; k < 128; ++k)
            s += hT[(size_t)b * 128 + k] * W1[(size_t)j * 128 + k];
        h[j] = s;
    }
    float mu = 0.0f;
    for (int j = 0; j < 64; ++j) mu += h[j];
    mu *= (1.0f / 64.0f);
    float var = 0.0f;
    for (int j = 0; j < 64; ++j) { float d = h[j] - mu; var += d * d; }
    var *= (1.0f / 64.0f);
    float inv = rsqrtf(var + 1e-5f);
    float s = b2[0];
    for (int j = 0; j < 64; ++j) {
        float v = (h[j] - mu) * inv * ln_g[j] + ln_b[j];
        v = (v >= 0.0f) ? v : 0.2f * v;
        s += v * W2[j];
    }
    out[b] = s;
}

// ---------------------------------------------------------------------------
extern "C" void kernel_launch(void* const* d_in, const int* in_sizes, int n_in,
                              void* d_out, int out_size, void* d_ws, size_t ws_size,
                              hipStream_t stream) {
    const float* map  = (const float*)d_in[0];
    const float* pos  = (const float*)d_in[1];
    const float* Wih0 = (const float*)d_in[2];
    const float* Whh0 = (const float*)d_in[3];
    const float* bih0 = (const float*)d_in[4];
    const float* bhh0 = (const float*)d_in[5];
    const float* Wih1 = (const float*)d_in[6];
    const float* Whh1 = (const float*)d_in[7];
    const float* bih1 = (const float*)d_in[8];
    const float* bhh1 = (const float*)d_in[9];
    const float* W1   = (const float*)d_in[10];
    const float* b1   = (const float*)d_in[11];
    const float* ln_g = (const float*)d_in[12];
    const float* ln_b = (const float*)d_in[13];
    const float* W2   = (const float*)d_in[14];
    const float* b2   = (const float*)d_in[15];

    char* ws = (char*)d_ws;
    size_t off = 0;
    unsigned short* xproj = (unsigned short*)(ws + off); off += (size_t)TB * G4H * 2;  // 134 MB
    unsigned short* x0    = (unsigned short*)(ws + off); off += (size_t)TB * 128 * 2;  // 33.5 MB
    unsigned short* hs0   = (unsigned short*)(ws + off); off += (size_t)TB * 128 * 2;  // 33.5 MB
    unsigned short* wih0b = (unsigned short*)(ws + off); off += (size_t)G4H * 128 * 2;
    unsigned short* whh0b = (unsigned short*)(ws + off); off += (size_t)G4H * 128 * 2;
    unsigned short* wih1b = (unsigned short*)(ws + off); off += (size_t)G4H * 128 * 2;
    unsigned short* whh1b = (unsigned short*)(ws + off); off += (size_t)G4H * 128 * 2;
    float*          bias0 = (float*)(ws + off);          off += (size_t)G4H * 4;
    float*          bias1 = (float*)(ws + off);          off += (size_t)G4H * 4;
    float*          hT1   = (float*)(ws + off);          off += (size_t)BB * 128 * 4;

    // --- prep: pack input + convert weights to bf16, fuse biases ---
    pack_x0_kernel<<<(TB * 128) / 256, 256, 0, stream>>>(map, pos, x0);
    cvt_bf16_kernel<<<(G4H * 128) / 256, 256, 0, stream>>>(Wih0, wih0b, G4H * 128);
    cvt_bf16_kernel<<<(G4H * 128) / 256, 256, 0, stream>>>(Whh0, whh0b, G4H * 128);
    cvt_bf16_kernel<<<(G4H * 128) / 256, 256, 0, stream>>>(Wih1, wih1b, G4H * 128);
    cvt_bf16_kernel<<<(G4H * 128) / 256, 256, 0, stream>>>(Whh1, whh1b, G4H * 128);
    bias_kernel<<<2, 256, 0, stream>>>(bih0, bhh0, bias0);
    bias_kernel<<<2, 256, 0, stream>>>(bih1, bhh1, bias1);

    // --- layer 0 ---
    const int gemm_blocks = (TB / 16) * 8 / 8;   // 8192
    xproj_gemm_kernel<<<gemm_blocks, 256, 0, stream>>>(x0, wih0b, bias0, xproj);
    lstm_rec_kernel<<<BB / 16, 256, 0, stream>>>(xproj, whh0b, hs0, nullptr);

    // --- layer 1 ---
    xproj_gemm_kernel<<<gemm_blocks, 256, 0, stream>>>(hs0, wih1b, bias1, xproj);
    lstm_rec_kernel<<<BB / 16, 256, 0, stream>>>(xproj, whh1b, nullptr, hT1);

    // --- head ---
    head_kernel<<<1, 64, 0, stream>>>(hT1, W1, b1, ln_g, ln_b, W2, b2,
                                      (float*)d_out);
}
